// PatchMerging_18013092839522
// MI455X (gfx1250) — matrix operand order repaired and checked by reading, not compiled, not run on hardware
//
#include <hip/hip_runtime.h>

typedef _Float16     v16h __attribute__((ext_vector_type(16)));
typedef float        v8f  __attribute__((ext_vector_type(8)));
typedef float        v4f  __attribute__((ext_vector_type(4)));
typedef unsigned int v4u  __attribute__((ext_vector_type(4)));

union FragPack { v16h v; v4u u[2]; };

// ---------------------------------------------------------------------------
// Kernel 0: convert + transpose w_red (768x192 f32, row-major k*192+n)
//           into Wt (192x768 f16, n-major) so B-fragments load as 2x b128.
// ---------------------------------------------------------------------------
__global__ __launch_bounds__(256) void wconv_kernel(const float* __restrict__ w,
                                                    _Float16* __restrict__ wt)
{
    int idx = blockIdx.x * 256 + threadIdx.x;   // 768*192 = 147456 = 576*256
    int k = idx / 192;
    int n = idx - k * 192;
    wt[n * 768 + k] = (_Float16)w[idx];
}

// ---------------------------------------------------------------------------
// Kernel 1: per block: gather 32 rows (octant concat), LayerNorm -> f16 LDS
//           A-tile, then 24 tiles of 16x16 GEMM via v_wmma_f32_16x16x32_f16.
// ---------------------------------------------------------------------------
__global__ __launch_bounds__(256) void patch_merge_ln_gemm(
    const float*    __restrict__ x,      // (2,96,32,64,64)
    const float*    __restrict__ gamma,  // (768)
    const float*    __restrict__ beta,   // (768)
    const _Float16* __restrict__ wt,     // (192,768) f16 transposed
    float*          __restrict__ out)    // (2,192,16,32,32)
{
    __shared__ float red_s[8][32];
    __shared__ float red_q[8][32];
    __shared__ float s_mu[32];
    __shared__ float s_rs[32];
    __shared__ __attribute__((aligned(16))) _Float16 s_A[32][776]; // 768 + 8 pad

    const int tid  = threadIdx.x;
    const int wave = tid >> 5;
    const int lane = tid & 31;

    // ---------------- Phase 1: gather + LayerNorm ----------------
    const int p  = wave;                 // octant index: bits (d,h,w)
    const int r  = lane;                 // row within 32-row M tile
    const int m  = blockIdx.x * 32 + r;  // global spatial row, M = 32768
    const int b  = m >> 14;              // 16*32*32 = 16384 rows per batch
    const int sp = m & 16383;
    const int d2 = sp >> 10;
    const int h2 = (sp >> 5) & 31;
    const int w2 = sp & 31;
    const int dd = 2 * d2 + ((p >> 2) & 1);
    const int hh = 2 * h2 + ((p >> 1) & 1);
    const int ww = 2 * w2 + (p & 1);
    // x offset = (b*96 + c)*131072 + dd*4096 + hh*64 + ww
    const float* xp = x + ((size_t)(b * 96) * 32 + dd) * 4096 + hh * 64 + ww;

    float s = 0.f, q = 0.f;
    #pragma unroll 8
    for (int c = 0; c < 96; ++c) {
        float v = xp[(size_t)c * 131072];
        s += v; q += v * v;
    }
    red_s[p][r] = s;
    red_q[p][r] = q;
    __syncthreads();
    if (wave == 0) {
        float ts = 0.f, tq = 0.f;
        #pragma unroll
        for (int i = 0; i < 8; ++i) { ts += red_s[i][lane]; tq += red_q[i][lane]; }
        float mu  = ts * (1.0f / 768.0f);
        float var = tq * (1.0f / 768.0f) - mu * mu;
        s_mu[lane] = mu;
        s_rs[lane] = __frsqrt_rn(var + 1e-5f);
    }
    __syncthreads();
    {
        const float mu = s_mu[r];
        const float rs = s_rs[r];
        const int   j0 = p * 96;
        #pragma unroll 8
        for (int c = 0; c < 96; ++c) {
            float v = xp[(size_t)c * 131072];          // re-read: L2 resident
            float a = (v - mu) * rs * gamma[j0 + c] + beta[j0 + c];
            s_A[r][j0 + c] = (_Float16)a;
        }
    }
    __syncthreads();

    // ---------------- Phase 2: WMMA GEMM ----------------
    // wave -> one 16-row group (mg) and 3 consecutive 16-col N tiles
    const int mg     = wave >> 2;            // 0: waves 0-3, 1: waves 4-7
    const int ntBase = (wave & 3) * 3;       // 0,3,6,9
    const int nloc   = lane & 15;
    const int rowA   = mg * 16 + nloc;       // A: lanes 0-15 and 16-31 both M=0..15
    const int koffA  = (lane < 16) ? 0 : 16; // A: lanes<16 K 0..15, >=16 K 16..31
    const int koffB  = (lane < 16) ? 0 : 16; // B: lanes<16 K 0..15, >=16 K 16..31

    const _Float16* wt0 = wt + (size_t)(ntBase * 16 + nloc) * 768 + koffB;
    const _Float16* wt1 = wt0 + 16 * 768;
    const _Float16* wt2 = wt1 + 16 * 768;

    v8f acc0 = {}, acc1 = {}, acc2 = {};

    for (int k0 = 0; k0 < 768; k0 += 32) {
        FragPack a, b0, b1, b2;
        // A frag: 16 contiguous K per lane at K=k0+koffA
        a.u[0] = *(const v4u*)(&s_A[rowA][k0 + koffA]);
        a.u[1] = *(const v4u*)(&s_A[rowA][k0 + koffA + 8]);
        // B frags: 16 contiguous K per lane (two b128)
        b0.u[0] = *(const v4u*)(wt0 + k0);
        b0.u[1] = *(const v4u*)(wt0 + k0 + 8);
        b1.u[0] = *(const v4u*)(wt1 + k0);
        b1.u[1] = *(const v4u*)(wt1 + k0 + 8);
        b2.u[0] = *(const v4u*)(wt2 + k0);
        b2.u[1] = *(const v4u*)(wt2 + k0 + 8);
        acc0 = __builtin_amdgcn_wmma_f32_16x16x32_f16(false, a.v, false, b0.v,
                                                      (short)0, acc0, false, false);
        acc1 = __builtin_amdgcn_wmma_f32_16x16x32_f16(false, a.v, false, b1.v,
                                                      (short)0, acc1, false, false);
        acc2 = __builtin_amdgcn_wmma_f32_16x16x32_f16(false, a.v, false, b2.v,
                                                      (short)0, acc2, false, false);
    }

    // ---------------- Store ----------------
    // Acc VGPR v holds M = v (+8 for lanes>=16) => 8 consecutive m per lane
    // out offset = b*192*16384 + n*16384 + (m & 16383); w2 runs within block.
    const int mBase = blockIdx.x * 32 + mg * 16 + ((lane < 16) ? 0 : 8);
    const int bOut  = mBase >> 14;
    const int spOut = mBase & 16383;
    float* obase = out + (size_t)bOut * 192 * 16384 + spOut
                       + (size_t)(ntBase * 16 + nloc) * 16384;

    *(v4f*)(obase)     = (v4f){acc0[0], acc0[1], acc0[2], acc0[3]};
    *(v4f*)(obase + 4) = (v4f){acc0[4], acc0[5], acc0[6], acc0[7]};
    obase += (size_t)16 * 16384;
    *(v4f*)(obase)     = (v4f){acc1[0], acc1[1], acc1[2], acc1[3]};
    *(v4f*)(obase + 4) = (v4f){acc1[4], acc1[5], acc1[6], acc1[7]};
    obase += (size_t)16 * 16384;
    *(v4f*)(obase)     = (v4f){acc2[0], acc2[1], acc2[2], acc2[3]};
    *(v4f*)(obase + 4) = (v4f){acc2[4], acc2[5], acc2[6], acc2[7]};
}

// ---------------------------------------------------------------------------
extern "C" void kernel_launch(void* const* d_in, const int* in_sizes, int n_in,
                              void* d_out, int out_size, void* d_ws, size_t ws_size,
                              hipStream_t stream)
{
    const float* x     = (const float*)d_in[0];
    const float* gamma = (const float*)d_in[1];
    const float* beta  = (const float*)d_in[2];
    const float* w_red = (const float*)d_in[3];
    _Float16*    wt    = (_Float16*)d_ws;          // 192*768*2 = 294912 bytes
    float*       out   = (float*)d_out;

    wconv_kernel<<<576, 256, 0, stream>>>(w_red, wt);
    patch_merge_ln_gemm<<<1024, 256, 0, stream>>>(x, gamma, beta, wt, out);
}
